// DecoderRNNT_70437463655146
// MI455X (gfx1250) — compile-verified, matrix operand order checked by reading, not run on hardware
//
#include <hip/hip_runtime.h>
#include <math.h>
#include <stdint.h>

#define B_      8
#define T_      128
#define U_      64
#define EPROJS_ 1024
#define ODIM_   1024
#define DUNITS_ 1024
#define EMBED_  512
#define JDIM_   640
#define H4_     4096

typedef __bf16   v8bf  __attribute__((ext_vector_type(8)));
typedef __bf16   v16bf __attribute__((ext_vector_type(16)));
typedef float    v8f   __attribute__((ext_vector_type(8)));
typedef uint32_t u32x4 __attribute__((ext_vector_type(4)));
typedef uint32_t u32x8 __attribute__((ext_vector_type(8)));

// ---------------------------------------------------------------------------
// WMMA helpers (wave32, 16x16x32 bf16 -> f32 accumulate).
// A-matrix 16x32 bf16 layout (ISA 7.12.2): lane L<16: M=L, K in {0..7}u{16..23};
// lane L>=16: M=L-16, K in {8..15}u{24..31}. B-matrix 32x16 is symmetric with
// N=lane%16, the lane's row K-contiguous -> two packed 16B loads per fragment.
// ---------------------------------------------------------------------------
__device__ __forceinline__ v16bf load_frag(const __bf16* rowptr, int kc) {
    int lane = threadIdx.x & 31;
    int koff = (lane & 16) ? 8 : 0;
    v8bf lo = *(const v8bf*)(rowptr + kc + koff);
    v8bf hi = *(const v8bf*)(rowptr + kc + koff + 16);
    v16bf a;
#pragma unroll
    for (int i = 0; i < 8; ++i) { a[i] = lo[i]; a[i + 8] = hi[i]; }
    return a;
}

__device__ __forceinline__ v8f wmma_bf16(v16bf a, v16bf b, v8f c) {
    return __builtin_amdgcn_wmma_f32_16x16x32_bf16(false, a, false, b, (short)0, c,
                                                   false, false);
}

__device__ __forceinline__ float sigm(float x) { return 1.0f / (1.0f + __expf(-x)); }

// ---------------------------------------------------------------------------
// Prep kernels
// ---------------------------------------------------------------------------
__global__ void k_convert_bf16(const float* __restrict__ src, __bf16* __restrict__ dst,
                               int n) {
    for (int i = blockIdx.x * blockDim.x + threadIdx.x; i < n; i += gridDim.x * blockDim.x)
        dst[i] = (__bf16)src[i];
}

// src: [4*1024][K] (gate-major, torch layout). dst row r = 4*unit+gate, bf16.
__global__ void k_permute_w(const float* __restrict__ src, __bf16* __restrict__ dst,
                            int K) {
    int n = H4_ * K;
    for (int i = blockIdx.x * blockDim.x + threadIdx.x; i < n; i += gridDim.x * blockDim.x) {
        int r = i / K, k = i - r * K;
        int unit = r >> 2, g = r & 3;
        dst[i] = (__bf16)src[(g * DUNITS_ + unit) * K + k];
    }
}

__global__ void k_bias_perm(const float* __restrict__ bi, const float* __restrict__ bh,
                            float* __restrict__ dst) {
    int i = blockIdx.x * blockDim.x + threadIdx.x;
    if (i < H4_) {
        int unit = i >> 2, g = i & 3;
        dst[i] = bi[g * DUNITS_ + unit] + bh[g * DUNITS_ + unit];
    }
}

// eys layout [U][16][512]: per-step contiguous 16-row tile, rows 8..15 zero.
__global__ void k_embed(const int* __restrict__ ys, const float* __restrict__ embed,
                        __bf16* __restrict__ eys) {
    int n = U_ * 16 * EMBED_;
    for (int i = blockIdx.x * blockDim.x + threadIdx.x; i < n; i += gridDim.x * blockDim.x) {
        int e = i & (EMBED_ - 1);
        int row = i >> 9;            // EMBED_ = 512
        int b = row & 15, u = row >> 4;
        eys[i] = (b < B_) ? (__bf16)embed[ys[b * U_ + u] * EMBED_ + e] : (__bf16)0.0f;
    }
}

// h buffers are 16 rows (rows 8..15 stay zero forever); c buffers 8 rows.
__global__ void k_zero_state(float* c0, float* c1, __bf16* h0, __bf16* h1) {
    int i = blockIdx.x * blockDim.x + threadIdx.x;
    if (i < 16 * DUNITS_) {
        h0[i] = (__bf16)0.0f; h1[i] = (__bf16)0.0f;
        if (i < B_ * DUNITS_) { c0[i] = 0.0f; c1[i] = 0.0f; }
    }
}

// ---------------------------------------------------------------------------
// One LSTM layer step, gates fused with c/h pointwise update.
// Gate-permuted weights: column 4*unit+gate -> a 16-wide N tile holds 4 whole
// units, so the owning wave finishes them in its epilogue (no extra kernel).
// grid: 64 blocks x 128 threads (4 waves); wave w owns N tile nt = bid*64+w*16.
// Two accumulator chains break the WMMA D->C RAW hazard chain.
// ---------------------------------------------------------------------------
__global__ __launch_bounds__(128)
void k_lstm_step(const __bf16* __restrict__ xA, int ldx, int Kx,   // [16][Kx], rows 8+ zero
                 const __bf16* __restrict__ hA,                    // [16][1024], rows 8+ zero
                 const __bf16* __restrict__ Wx,                    // [4096][Kx]  permuted
                 const __bf16* __restrict__ Wh,                    // [4096][1024] permuted
                 const float* __restrict__ bias,                   // [4096] permuted
                 float* __restrict__ cstate,                       // [8][1024]
                 __bf16* __restrict__ hout,                        // [16][1024]
                 __bf16* __restrict__ hdec, int ld_hdec) {
    int lane = threadIdx.x & 31;
    int wave = threadIdx.x >> 5;
    int nt = blockIdx.x * 64 + wave * 16;
    int mrow = lane & 15;
    int ncol = nt + (lane & 15);

    v8f acc0 = {}, acc1 = {};
    const __bf16* arow = xA + mrow * ldx;
    const __bf16* brow = Wx + ncol * Kx;
    for (int kc = 0; kc < Kx; kc += 64) {
        acc0 = wmma_bf16(load_frag(arow, kc),      load_frag(brow, kc),      acc0);
        acc1 = wmma_bf16(load_frag(arow, kc + 32), load_frag(brow, kc + 32), acc1);
    }
    arow = hA + mrow * DUNITS_;
    brow = Wh + ncol * DUNITS_;
    for (int kc = 0; kc < DUNITS_; kc += 64) {
        acc0 = wmma_bf16(load_frag(arow, kc),      load_frag(brow, kc),      acc0);
        acc1 = wmma_bf16(load_frag(arow, kc + 32), load_frag(brow, kc + 32), acc1);
    }
#pragma unroll
    for (int r = 0; r < 8; ++r) acc0[r] += acc1[r];

    // dump the 16x16 gate tile to LDS, then do the pointwise cell update
    __shared__ float sC[4][16][17];
#pragma unroll
    for (int r = 0; r < 8; ++r)
        sC[wave][r + ((lane >> 4) << 3)][lane & 15] = acc0[r];
    __syncthreads();

    int m = lane >> 2;           // batch row 0..7
    int q = lane & 3;            // local unit 0..3
    float gi = sC[wave][m][4 * q + 0] + bias[nt + 4 * q + 0];
    float gf = sC[wave][m][4 * q + 1] + bias[nt + 4 * q + 1];
    float gg = sC[wave][m][4 * q + 2] + bias[nt + 4 * q + 2];
    float go = sC[wave][m][4 * q + 3] + bias[nt + 4 * q + 3];
    int unit = (nt >> 2) + q;
    int ci = m * DUNITS_ + unit;
    float cn = sigm(gf) * cstate[ci] + sigm(gi) * tanhf(gg);
    float hn = sigm(go) * tanhf(cn);
    cstate[ci] = cn;
    hout[ci] = (__bf16)hn;
    if (hdec) hdec[m * ld_hdec + unit] = (__bf16)hn;
}

// ---------------------------------------------------------------------------
// C[M,N] = A[M,K] @ Bw[N,K]^T (+bias). M, N multiples of 64 -> no guards.
// grid (M/64, N/64), 256 threads (8 waves); each wave: 1 M-tile x 2 N-tiles.
// ---------------------------------------------------------------------------
__global__ __launch_bounds__(256)
void k_gemm_bias(const __bf16* __restrict__ A, int lda,
                 const __bf16* __restrict__ Bw, int ldb,
                 const float* __restrict__ bias,
                 float* __restrict__ C, int ldc, int K) {
    int lane = threadIdx.x & 31;
    int wave = threadIdx.x >> 5;
    int mbase = blockIdx.x * 64 + (wave >> 1) * 16;
    int nbase = blockIdx.y * 64 + (wave & 1) * 32;

    const __bf16* arow  = A + (mbase + (lane & 15)) * lda;
    const __bf16* brow0 = Bw + (nbase + (lane & 15)) * ldb;
    const __bf16* brow1 = brow0 + 16 * ldb;

    v8f c0 = {}, c1 = {};
    for (int kc = 0; kc < K; kc += 32) {
        __builtin_prefetch(brow0 + kc + 256, 0, 1);
        v16bf a = load_frag(arow, kc);
        c0 = wmma_bf16(a, load_frag(brow0, kc), c0);
        c1 = wmma_bf16(a, load_frag(brow1, kc), c1);
    }
    int n0 = nbase + (lane & 15);
    float bv0 = bias ? bias[n0] : 0.0f;
    float bv1 = bias ? bias[n0 + 16] : 0.0f;
#pragma unroll
    for (int r = 0; r < 8; ++r) {
        int mm = mbase + r + ((lane >> 4) << 3);
        C[mm * ldc + n0]      = c0[r] + bv0;
        C[mm * ldc + n0 + 16] = c1[r] + bv1;
    }
}

// ---------------------------------------------------------------------------
// Stage A = tanh(ze[b,t,:] + zd[b,u,:]) once, as bf16 rows (K-contiguous).
// 80 MB result fits in L2, so the joint GEMM re-reads it from L2, and the
// 42M transcendentals are computed exactly once (vs once per N-block).
// grid (B*T), 256 threads.
// ---------------------------------------------------------------------------
__global__ __launch_bounds__(256)
void k_stage(const float* __restrict__ ze,        // [B*T][640]
             const float* __restrict__ zd,        // [B*U][640]
             __bf16* __restrict__ tA) {           // [B*T*U][640]
    __shared__ float sZE[JDIM_];
    int bt = blockIdx.x;
    int b  = bt >> 7;                             // T_ = 128
    for (int j = threadIdx.x; j < JDIM_; j += 256) sZE[j] = ze[bt * JDIM_ + j];
    __syncthreads();
    for (int u = 0; u < U_; ++u) {
        const float* zr = zd + (b * U_ + u) * JDIM_;
        __bf16* dst = tA + ((size_t)bt * U_ + u) * JDIM_;
        for (int j = threadIdx.x; j < JDIM_; j += 256)
            dst[j] = (__bf16)tanhf(sZE[j] + zr[j]);
    }
}

// ---------------------------------------------------------------------------
// Joint output GEMM: out[row, o] = A[row,:] @ Wout[o,:] + b_out[o],
// row = (b*T+t)*U+u, 65536 x 1024 x 640 in bf16 WMMA.
// Per WG (64 rows x 128 cols):
//   - TDM iterate-mode D# streams the 128x640 bf16 W block into LDS with a
//     648-element row stride (banks 4n mod 64 -> conflict-free ds_load_b128),
//     overlapped with global_prefetch of the A rows; s_wait_tensorcnt + barrier.
//   - 20 k-steps: 1 A fragment (global, L2-resident) amortized over 4 N-tiles.
// grid (65536/64 = 1024, 1024/128 = 8), 256 threads (8 waves x 4 tiles).
// ---------------------------------------------------------------------------
__global__ __launch_bounds__(256)
void k_joint(const __bf16* __restrict__ A,        // [65536][640] tanh'd bf16
             const __bf16* __restrict__ Wout,     // [1024][640] bf16
             const float* __restrict__ bout,      // [1024]
             float* __restrict__ out) {           // [B,T,U,1024]
    __shared__ __bf16 sW[128][648];               // padded rows: bank-conflict-free

    int mb = blockIdx.x * 64;
    int nbase = blockIdx.y * 128;

    // ---- TDM load of W block (wave 0), iterate-mode D# (ISA 08 sec 8/9.3) ----
    {
        uint64_t ga  = (uint64_t)(uintptr_t)(Wout + (size_t)nbase * JDIM_);
        uint32_t lds = (uint32_t)(uintptr_t)(&sW[0][0]);  // LDS aperture: addr[31:0]
        u32x4 g0;
        g0[0] = 1u;                                  // count=1 (valid user D#)
        g0[1] = lds;                                 // lds_addr
        g0[2] = (uint32_t)ga;                        // global_addr[31:0]
        g0[3] = (uint32_t)(ga >> 32) | 0x80000000u;  // global_addr[56:32] | type=2
        u32x8 g1;
        g1[0] = (1u << 16) | (1u << 19);             // data_size=2B | iterate_enable
        g1[1] = (uint32_t)JDIM_ << 16;               // tensor_dim0 = 640
        g1[2] = 1u << 16;                            // tensor_dim1 = 1
        g1[3] = (uint32_t)JDIM_ << 16;               // tile_dim0 = 640
        g1[4] = 1u;                                  // tile_dim1 = 1 (one row/iter)
        g1[5] = (uint32_t)JDIM_;                     // tensor_dim0_stride = 640
        g1[6] = (uint32_t)JDIM_ << 16;               // tensor_dim1_stride[15:0]
        g1[7] = 0u;
        u32x4 g2;
        g2[0] = 0u;                                  // tensor_dim2 (unused)
        g2[1] = 648u;                                // lds_addr_increment (elements)
        g2[2] = (uint32_t)JDIM_;                     // global_addr_increment (elements)
        g2[3] = 127u << 16;                          // iterate_count: 128 iterations
        u32x4 g3;
        g3[0] = 0u; g3[1] = 0u; g3[2] = 0u; g3[3] = 0u;
        if (threadIdx.x < 32)
            asm volatile("tensor_load_to_lds %0, %1, %2, %3"
                         :: "s"(g0), "s"(g1), "s"(g2), "s"(g3) : "memory");
    }

    int lane = threadIdx.x & 31;
    int wave = threadIdx.x >> 5;
    int mt = wave >> 1;                           // M tile 0..3
    int nloc = (wave & 1) * 64;                   // local N base (0 or 64)

    const __bf16* arow = A + (size_t)(mb + mt * 16 + (lane & 15)) * JDIM_;
    // prefetch this lane's A row while the TDM streams W into LDS
    for (int kc = 0; kc < JDIM_; kc += 64)
        __builtin_prefetch(arow + kc, 0, 1);

    if (threadIdx.x < 32) __builtin_amdgcn_s_wait_tensorcnt(0);
    __syncthreads();

    const __bf16* b0 = &sW[nloc + (lane & 15)][0];
    const __bf16* b1 = b0 + 16 * 648;
    const __bf16* b2 = b0 + 32 * 648;
    const __bf16* b3 = b0 + 48 * 648;

    v8f c0 = {}, c1 = {}, c2 = {}, c3 = {};
    for (int kc = 0; kc < JDIM_; kc += 32) {
        v16bf a = load_frag(arow, kc);
        c0 = wmma_bf16(a, load_frag(b0, kc), c0);
        c1 = wmma_bf16(a, load_frag(b1, kc), c1);
        c2 = wmma_bf16(a, load_frag(b2, kc), c2);
        c3 = wmma_bf16(a, load_frag(b3, kc), c3);
    }

    int n0 = nbase + nloc + (lane & 15);
    float bv0 = bout[n0], bv1 = bout[n0 + 16], bv2 = bout[n0 + 32], bv3 = bout[n0 + 48];
#pragma unroll
    for (int r = 0; r < 8; ++r) {
        long row = mb + mt * 16 + r + ((lane >> 4) << 3);
        long o = row * (long)ODIM_;
        out[o + n0]      = c0[r] + bv0;
        out[o + n0 + 16] = c1[r] + bv1;
        out[o + n0 + 32] = c2[r] + bv2;
        out[o + n0 + 48] = c3[r] + bv3;
    }
}

// ---------------------------------------------------------------------------
extern "C" void kernel_launch(void* const* d_in, const int* in_sizes, int n_in,
                              void* d_out, int out_size, void* d_ws, size_t ws_size,
                              hipStream_t stream) {
    (void)in_sizes; (void)n_in; (void)out_size; (void)ws_size;
    const float* hs_pad = (const float*)d_in[0];
    const int*   ys     = (const int*)d_in[1];
    const float* embed  = (const float*)d_in[2];
    const float* W_ih0  = (const float*)d_in[3];
    const float* W_hh0  = (const float*)d_in[4];
    const float* b_ih0  = (const float*)d_in[5];
    const float* b_hh0  = (const float*)d_in[6];
    const float* W_ih1  = (const float*)d_in[7];
    const float* W_hh1  = (const float*)d_in[8];
    const float* b_ih1  = (const float*)d_in[9];
    const float* b_hh1  = (const float*)d_in[10];
    const float* W_enc  = (const float*)d_in[11];
    const float* b_enc  = (const float*)d_in[12];
    const float* W_dec  = (const float*)d_in[13];
    const float* W_out  = (const float*)d_in[14];
    const float* b_out  = (const float*)d_in[15];
    float* out = (float*)d_out;

    char* ws = (char*)d_ws;
    size_t off = 0;
    auto alloc = [&](size_t bytes) -> char* {
        size_t o = (off + 255) & ~(size_t)255;
        off = o + bytes;
        return ws + o;
    };

    __bf16* eys     = (__bf16*)alloc((size_t)U_ * 16 * EMBED_ * 2);   // [U][16][512]
    __bf16* Wih0p   = (__bf16*)alloc((size_t)H4_ * EMBED_ * 2);
    __bf16* Whh0p   = (__bf16*)alloc((size_t)H4_ * DUNITS_ * 2);
    __bf16* Wih1p   = (__bf16*)alloc((size_t)H4_ * DUNITS_ * 2);
    __bf16* Whh1p   = (__bf16*)alloc((size_t)H4_ * DUNITS_ * 2);
    float*  bias0p  = (float*)alloc((size_t)H4_ * 4);
    float*  bias1p  = (float*)alloc((size_t)H4_ * 4);
    float*  c0      = (float*)alloc((size_t)B_ * DUNITS_ * 4);
    float*  c1      = (float*)alloc((size_t)B_ * DUNITS_ * 4);
    __bf16* h0      = (__bf16*)alloc((size_t)16 * DUNITS_ * 2);       // 16 rows, 8+ zero
    __bf16* h1      = (__bf16*)alloc((size_t)16 * DUNITS_ * 2);
    __bf16* hdec    = (__bf16*)alloc((size_t)B_ * U_ * DUNITS_ * 2);
    __bf16* hs_bf   = (__bf16*)alloc((size_t)B_ * T_ * EPROJS_ * 2);
    __bf16* Wenc_bf = (__bf16*)alloc((size_t)JDIM_ * EPROJS_ * 2);
    __bf16* Wdec_bf = (__bf16*)alloc((size_t)JDIM_ * DUNITS_ * 2);
    __bf16* Wout_bf = (__bf16*)alloc((size_t)ODIM_ * JDIM_ * 2);
    float*  ze      = (float*)alloc((size_t)B_ * T_ * JDIM_ * 4);
    float*  zd      = (float*)alloc((size_t)B_ * U_ * JDIM_ * 4);
    __bf16* tA      = (__bf16*)alloc((size_t)B_ * T_ * U_ * JDIM_ * 2); // 80 MB, L2-resident

    const int TB = 256;
    auto grd = [](int n, int tb) { return dim3((unsigned)((n + tb - 1) / tb)); };

    // ---- one-time prep (bf16 conversion / gate permutation / gather) ----
    k_convert_bf16<<<grd(B_ * T_ * EPROJS_, TB), TB, 0, stream>>>(hs_pad, hs_bf, B_ * T_ * EPROJS_);
    k_convert_bf16<<<grd(JDIM_ * EPROJS_, TB), TB, 0, stream>>>(W_enc, Wenc_bf, JDIM_ * EPROJS_);
    k_convert_bf16<<<grd(JDIM_ * DUNITS_, TB), TB, 0, stream>>>(W_dec, Wdec_bf, JDIM_ * DUNITS_);
    k_convert_bf16<<<grd(ODIM_ * JDIM_, TB), TB, 0, stream>>>(W_out, Wout_bf, ODIM_ * JDIM_);
    k_permute_w<<<grd(H4_ * EMBED_, TB), TB, 0, stream>>>(W_ih0, Wih0p, EMBED_);
    k_permute_w<<<grd(H4_ * DUNITS_, TB), TB, 0, stream>>>(W_hh0, Whh0p, DUNITS_);
    k_permute_w<<<grd(H4_ * DUNITS_, TB), TB, 0, stream>>>(W_ih1, Wih1p, DUNITS_);
    k_permute_w<<<grd(H4_ * DUNITS_, TB), TB, 0, stream>>>(W_hh1, Whh1p, DUNITS_);
    k_bias_perm<<<grd(H4_, TB), TB, 0, stream>>>(b_ih0, b_hh0, bias0p);
    k_bias_perm<<<grd(H4_, TB), TB, 0, stream>>>(b_ih1, b_hh1, bias1p);
    k_embed<<<grd(U_ * 16 * EMBED_, TB), TB, 0, stream>>>(ys, embed, eys);
    k_zero_state<<<grd(16 * DUNITS_, TB), TB, 0, stream>>>(c0, c1, h0, h1);

    // ---- sequential 2-layer LSTM scan (fused gates + pointwise) ----
    for (int u = 0; u < U_; ++u) {
        k_lstm_step<<<64, 128, 0, stream>>>(eys + (size_t)u * 16 * EMBED_, EMBED_, EMBED_,
                                            h0, Wih0p, Whh0p, bias0p, c0, h0,
                                            (__bf16*)nullptr, 0);
        k_lstm_step<<<64, 128, 0, stream>>>(h0, DUNITS_, DUNITS_,
                                            h1, Wih1p, Whh1p, bias1p, c1, h1,
                                            hdec + (size_t)u * DUNITS_, U_ * DUNITS_);
    }

    // ---- ze / zd projections ----
    k_gemm_bias<<<dim3(B_ * T_ / 64, JDIM_ / 64), 256, 0, stream>>>(
        hs_bf, EPROJS_, Wenc_bf, EPROJS_, b_enc, ze, JDIM_, EPROJS_);
    k_gemm_bias<<<dim3(B_ * U_ / 64, JDIM_ / 64), 256, 0, stream>>>(
        hdec, DUNITS_, Wdec_bf, DUNITS_, (const float*)nullptr, zd, JDIM_, DUNITS_);

    // ---- tanh staging (once) then the 256 MB / 86 GFLOP output GEMM ----
    k_stage<<<dim3(B_ * T_), 256, 0, stream>>>(ze, zd, tA);
    k_joint<<<dim3(B_ * T_ * U_ / 64, ODIM_ / 128), 256, 0, stream>>>(tA, Wout_bf, b_out, out);
}
